// DGGCN_37409165148374
// MI455X (gfx1250) — compile-verified, matrix operand order checked by reading, not run on hardware
//
#include <hip/hip_runtime.h>
#include <hip/hip_bf16.h>
#include <math.h>

// Problem constants
#define NB 16
#define NC 256
#define NT 128
#define NV 64
#define NK 8
#define NO 256
#define NCB 32          // NO / NK
#define NTV (NT*NV)     // 8192

typedef __attribute__((ext_vector_type(16))) __bf16 v16bf;
typedef __attribute__((ext_vector_type(8)))  float  v8f;
typedef __attribute__((ext_vector_type(4)))  unsigned int u32x4;

union FragU { v16bf v; u32x4 q[2]; };

__device__ __forceinline__ unsigned short f2bfu(float f) {
  union { float f; unsigned u; } a; a.f = f;
  unsigned u = a.u;
  return (unsigned short)((u + 0x7FFFu + ((u >> 16) & 1u)) >> 16); // RNE
}
__device__ __forceinline__ __bf16 f2bf(float f) {
  union { unsigned short s; __bf16 h; } o; o.s = f2bfu(f);
  return o.h;
}
__device__ __forceinline__ __bf16 u2bf(unsigned short s) {
  union { unsigned short s; __bf16 h; } o; o.s = s;
  return o.h;
}
__device__ __forceinline__ float bf2f(unsigned short s) {
  union { unsigned u; float f; } o; o.u = ((unsigned)s) << 16;
  return o.f;
}

// CDNA5 async global->LDS copy (16B per lane), tracked by ASYNCcnt.
__device__ __forceinline__ void async_ld_b128(unsigned lds_off,
                                              unsigned long long gaddr) {
  asm volatile("global_load_async_to_lds_b128 %0, %1, off"
               :: "v"(lds_off), "v"(gaddr) : "memory");
}
__device__ __forceinline__ void wait_async0() {
  asm volatile("s_wait_asynccnt 0x0" ::: "memory");
}

// ---------------------------------------------------------------------------
// K0: one-time fp32 -> bf16 weight conversion
__global__ void k_w2bf(const float* __restrict__ src,
                       unsigned short* __restrict__ dst, int n) {
  int i = blockIdx.x * 256 + threadIdx.x;
  if (i < n) dst[i] = f2bfu(src[i]);
}

// ---------------------------------------------------------------------------
// K1: tp[b][c][v] = mean_t x[b][c][t][v]
__global__ void k_tp(const float* __restrict__ x, float* __restrict__ tp) {
  int bc = blockIdx.x;            // b*NC + c
  int v  = threadIdx.x;           // 64 threads
  const float* p = x + (size_t)bc * NTV + v;
  float s = 0.f;
  #pragma unroll 8
  for (int t = 0; t < NT; ++t) s += p[t * NV];
  tp[(size_t)bc * NV + v] = s * (1.0f / NT);
}

// ---------------------------------------------------------------------------
// K2: x1[b][o][v], x2[b][o][v]  (fp32 dot over C; sensitive path, keep exact)
__global__ __launch_bounds__(256) void k_x1x2(
    const float* __restrict__ tp, const float* __restrict__ w1,
    const float* __restrict__ b1, const float* __restrict__ w2,
    const float* __restrict__ b2, float* __restrict__ x1,
    float* __restrict__ x2) {
  int i = blockIdx.x * 256 + threadIdx.x;        // < NB*NO*NV
  int v = i & 63;
  int o = (i >> 6) & 255;
  int b = i >> 14;
  const float* tpr = tp + (size_t)b * NC * NV + v;
  const float* w1r = w1 + (size_t)o * NC;
  const float* w2r = w2 + (size_t)o * NC;
  float s1 = 0.f, s2 = 0.f;
  #pragma unroll 8
  for (int cc = 0; cc < NC; ++cc) {
    float t = tpr[cc * NV];
    s1 += w1r[cc] * t;
    s2 += w2r[cc] * t;
  }
  x1[i] = s1 + b1[o];
  x2[i] = s2 + b2[o];
}

// ---------------------------------------------------------------------------
// K3: ada[b][k][v][w] = softmax_v( sum_c x1[b,k,c,v]*x2[b,k,c,w] )
__global__ __launch_bounds__(256) void k_ada(
    const float* __restrict__ x1, const float* __restrict__ x2,
    float* __restrict__ ada) {
  int bk = blockIdx.x;            // b*NK + k
  int b = bk >> 3, k = bk & 7;
  __shared__ float S[NV * NV];
  const float* x1b = x1 + ((size_t)b * NO + k * NCB) * NV;   // [32][64]
  const float* x2b = x2 + ((size_t)b * NO + k * NCB) * NV;
  #pragma unroll
  for (int i = 0; i < 16; ++i) {
    int e = threadIdx.x + 256 * i;         // v*64+w
    int v = e >> 6, w = e & 63;
    float s = 0.f;
    #pragma unroll
    for (int cc = 0; cc < NCB; ++cc) s += x1b[cc * NV + v] * x2b[cc * NV + w];
    S[e] = s;
  }
  __syncthreads();
  if (threadIdx.x < NV) {
    int w = threadIdx.x;
    float mx = -1e30f;
    for (int v = 0; v < NV; ++v) mx = fmaxf(mx, S[v * NV + w]);
    float sum = 0.f;
    for (int v = 0; v < NV; ++v) sum += __expf(S[v * NV + w] - mx);
    float inv = 1.f / sum;
    float* ar = ada + (size_t)bk * NV * NV;
    for (int v = 0; v < NV; ++v) ar[v * NV + w] = __expf(S[v * NV + w] - mx) * inv;
  }
}

// ---------------------------------------------------------------------------
// K4/K8: per-batch GEMM  Y[b][m][n] = sum_k W[m][k]*X[b][k][n] + bias[m]
// M=K=256, N=8192. Tile 128x128, BK=32, bf16 WMMA with f32 accumulate.
// A tile: async global->LDS copy of pre-converted bf16 weights.
// BF16IN: X is bf16 (head path);  BF16OUT: Y stored as bf16 (stem path).
template<bool BF16IN, bool BF16OUT>
__global__ __launch_bounds__(256) void k_gemm256(
    const unsigned short* __restrict__ Wbf,   // [256][256] bf16 row-major
    const float* __restrict__ bias,
    const void* __restrict__ Xv,              // [B][256][8192]
    void* __restrict__ Yv) {                  // [B][256][8192]
  const int n0 = blockIdx.x * 128;
  const int m0 = blockIdx.y * 128;
  const int b  = blockIdx.z;
  const int tid  = threadIdx.x;
  const int wave = tid >> 5;
  const int lane = tid & 31;
  const int lr = lane & 15;       // row within 16
  const int g  = lane >> 4;       // half select

  __shared__ __align__(16) __bf16 As[128 * 32];   // [m][k]
  __shared__ __align__(16) __bf16 Bs[128 * 32];   // [n][k] (transposed)

  v8f acc[8];
  v8f zero = {};
  #pragma unroll
  for (int i = 0; i < 8; ++i) acc[i] = zero;

  const unsigned lds_as = (unsigned)(uintptr_t)&As[0];

  for (int k0 = 0; k0 < 256; k0 += 32) {
    // --- A tile: 128x32 bf16, layout-preserving async copy (8KB, 512x16B)
    #pragma unroll
    for (int i = 0; i < 2; ++i) {
      int e = tid + 256 * i;                 // 512 chunks
      int r = e >> 2, ch = e & 3;            // row, 16B chunk in row
      unsigned long long ga = (unsigned long long)(uintptr_t)
          (Wbf + (size_t)(m0 + r) * 256 + k0 + ch * 8);
      unsigned la = lds_as + (unsigned)(r * 64 + ch * 16);
      async_ld_b128(la, ga);
    }
    // --- B tile: 32x128 -> LDS [n][k]
    if constexpr (BF16IN) {
      const unsigned short* Xb = (const unsigned short*)Xv + (size_t)b * NC * NTV;
      #pragma unroll
      for (int i = 0; i < 16; ++i) {
        int e = tid + 256 * i;
        int kk = e >> 7, n = e & 127;
        Bs[n * 32 + kk] = u2bf(Xb[(size_t)(k0 + kk) * NTV + n0 + n]);
      }
      if (k0 < 224)   // prefetch next K-step rows
        __builtin_prefetch(Xb + (size_t)(k0 + 32 + (tid >> 4)) * NTV + n0 + (tid & 15) * 8, 0, 1);
    } else {
      const float* Xb = (const float*)Xv + (size_t)b * NC * NTV;
      #pragma unroll
      for (int i = 0; i < 16; ++i) {
        int e = tid + 256 * i;
        int kk = e >> 7, n = e & 127;
        Bs[n * 32 + kk] = f2bf(Xb[(size_t)(k0 + kk) * NTV + n0 + n]);
      }
      if (k0 < 224)
        __builtin_prefetch(Xb + (size_t)(k0 + 32 + (tid >> 4)) * NTV + n0 + (tid & 15) * 8, 0, 1);
    }
    wait_async0();
    __syncthreads();

    const unsigned* arow = (const unsigned*)&As[(wave * 16 + lr) * 32];
    FragU af;
    af.q[0] = *(const u32x4*)(arow + g * 4);
    af.q[1] = *(const u32x4*)(arow + 8 + g * 4);
    #pragma unroll
    for (int nt = 0; nt < 8; ++nt) {
      const unsigned* brow = (const unsigned*)&Bs[(nt * 16 + lr) * 32];
      FragU bf;
      bf.q[0] = *(const u32x4*)(brow + g * 4);
      bf.q[1] = *(const u32x4*)(brow + 8 + g * 4);
      acc[nt] = __builtin_amdgcn_wmma_f32_16x16x32_bf16(
          false, af.v, false, bf.v, (short)0, acc[nt], false, false);
    }
    __syncthreads();
  }

  const int mbase = m0 + wave * 16;
  #pragma unroll
  for (int nt = 0; nt < 8; ++nt) {
    int n = n0 + nt * 16 + lr;
    #pragma unroll
    for (int r = 0; r < 8; ++r) {
      int m = mbase + r + 8 * g;
      float val = acc[nt][r] + bias[m];
      if constexpr (BF16OUT) {
        ((unsigned short*)Yv)[((size_t)b * NO + m) * NTV + n] = f2bfu(val);
      } else {
        ((float*)Yv)[((size_t)b * NO + m) * NTV + n] = val;
      }
    }
  }
}

// ---------------------------------------------------------------------------
// K5/K9: per-(b,o) partial sum / sumsq (deterministic two-stage BN stats)
__global__ __launch_bounds__(256) void k_chan_partial_f32(
    const float* __restrict__ y, float* __restrict__ partial) {
  int bo = blockIdx.x;            // b*NO + o
  int b = bo >> 8, o = bo & 255;
  const float* p = y + (size_t)bo * NTV;
  float s = 0.f, sq = 0.f;
  for (int i = threadIdx.x; i < NTV; i += 256) {
    float v = p[i];
    s += v; sq += v * v;
  }
  __shared__ float sh[512];
  sh[threadIdx.x] = s;
  sh[256 + threadIdx.x] = sq;
  __syncthreads();
  for (int st = 128; st > 0; st >>= 1) {
    if (threadIdx.x < st) {
      sh[threadIdx.x] += sh[threadIdx.x + st];
      sh[256 + threadIdx.x] += sh[256 + threadIdx.x + st];
    }
    __syncthreads();
  }
  if (threadIdx.x == 0) {
    partial[(size_t)(o * NB + b) * 2 + 0] = sh[0];
    partial[(size_t)(o * NB + b) * 2 + 1] = sh[256];
  }
}

__global__ __launch_bounds__(256) void k_chan_partial_bf16(
    const unsigned short* __restrict__ y, float* __restrict__ partial) {
  int bo = blockIdx.x;
  int b = bo >> 8, o = bo & 255;
  const unsigned short* p = y + (size_t)bo * NTV;
  float s = 0.f, sq = 0.f;
  for (int i = threadIdx.x; i < NTV; i += 256) {
    float v = bf2f(p[i]);
    s += v; sq += v * v;
  }
  __shared__ float sh[512];
  sh[threadIdx.x] = s;
  sh[256 + threadIdx.x] = sq;
  __syncthreads();
  for (int st = 128; st > 0; st >>= 1) {
    if (threadIdx.x < st) {
      sh[threadIdx.x] += sh[threadIdx.x + st];
      sh[256 + threadIdx.x] += sh[256 + threadIdx.x + st];
    }
    __syncthreads();
  }
  if (threadIdx.x == 0) {
    partial[(size_t)(o * NB + b) * 2 + 0] = sh[0];
    partial[(size_t)(o * NB + b) * 2 + 1] = sh[256];
  }
}

// K6/K10: finalize per-channel scale/shift
__global__ void k_chan_final(const float* __restrict__ partial,
                             const float* __restrict__ gamma,
                             const float* __restrict__ beta,
                             float* __restrict__ coef) {
  int o = threadIdx.x;            // 256 threads
  float s = 0.f, sq = 0.f;
  for (int b = 0; b < NB; ++b) {
    s  += partial[(size_t)(o * NB + b) * 2 + 0];
    sq += partial[(size_t)(o * NB + b) * 2 + 1];
  }
  const float Nf = (float)NB * NT * NV;
  float mean = s / Nf;
  float var  = sq / Nf - mean * mean;
  float inv  = rsqrtf(var + 1e-5f);
  float scale = gamma[o] * inv;
  coef[o * 2 + 0] = scale;
  coef[o * 2 + 1] = beta[o] - mean * scale;
}

// ---------------------------------------------------------------------------
// K7: fused BN+ReLU, on-the-fly A-tile build, 128x64x64 WMMA per (b,k,c)
__global__ __launch_bounds__(256) void k_einsum(
    const unsigned short* __restrict__ Yst,  // bf16 stem activations
    const float* __restrict__ coef,
    const float* __restrict__ Ap, const float* __restrict__ x1,
    const float* __restrict__ x2, const float* __restrict__ ada,
    const float* __restrict__ alpha_p, const float* __restrict__ beta_p,
    unsigned short* __restrict__ H2) {       // bf16 out
  int idx = blockIdx.x;                 // b*256 + k*32 + c
  int c = idx & 31, k = (idx >> 5) & 7, b = idx >> 8;
  int o = k * NCB + c;
  const int tid = threadIdx.x, wave = tid >> 5, lane = tid & 31;
  const int lr = lane & 15, g = lane >> 4;
  const float alpha = alpha_p[0], beta = beta_p[0];

  __shared__ __align__(16) __bf16 Ah[NT * NV];    // [t][v]  16KB
  __shared__ __align__(16) __bf16 Bt[NV * NV];    // [w][v]   8KB

  float scale = coef[o * 2], shift = coef[o * 2 + 1];
  const unsigned short* yrow = Yst + ((size_t)b * NO + o) * NTV;
  #pragma unroll
  for (int i = 0; i < 32; ++i) {
    int e = tid + 256 * i;                        // < 8192
    float v = bf2f(yrow[e]) * scale + shift;
    Ah[e] = f2bf(v > 0.f ? v : 0.f);
  }
  const float* x1r = x1 + ((size_t)b * NO + o) * NV;
  const float* x2r = x2 + ((size_t)b * NO + o) * NV;
  const float* adar = ada + (size_t)(b * NK + k) * NV * NV;
  const float* apr  = Ap  + (size_t)k * NV * NV;
  #pragma unroll
  for (int i = 0; i < 16; ++i) {
    int e = tid + 256 * i;                        // v*64 + w
    int v = e >> 6, w = e & 63;
    float a = apr[e] + alpha * tanhf(x1r[v] - x2r[w]) + beta * adar[e];
    Bt[w * NV + v] = f2bf(a);                     // K(=v)-major per n row
  }
  __syncthreads();

  v8f acc[4];
  v8f zero = {};
  #pragma unroll
  for (int i = 0; i < 4; ++i) acc[i] = zero;

  #pragma unroll
  for (int s = 0; s < 2; ++s) {                   // K = 64 in two steps of 32
    const unsigned* arow = (const unsigned*)&Ah[(wave * 16 + lr) * NV];
    FragU af;
    af.q[0] = *(const u32x4*)(arow + s * 16 + g * 4);
    af.q[1] = *(const u32x4*)(arow + s * 16 + 8 + g * 4);
    #pragma unroll
    for (int nt = 0; nt < 4; ++nt) {
      const unsigned* brow = (const unsigned*)&Bt[(nt * 16 + lr) * NV];
      FragU bf;
      bf.q[0] = *(const u32x4*)(brow + s * 16 + g * 4);
      bf.q[1] = *(const u32x4*)(brow + s * 16 + 8 + g * 4);
      acc[nt] = __builtin_amdgcn_wmma_f32_16x16x32_bf16(
          false, af.v, false, bf.v, (short)0, acc[nt], false, false);
    }
  }

  unsigned short* out = H2 + ((size_t)b * NO + o) * NTV;
  #pragma unroll
  for (int nt = 0; nt < 4; ++nt) {
    int w = nt * 16 + lr;
    #pragma unroll
    for (int r = 0; r < 8; ++r) {
      int t = wave * 16 + r + 8 * g;
      out[t * NV + w] = f2bfu(acc[nt][r]);
    }
  }
}

// ---------------------------------------------------------------------------
// K11: out = relu(bn(z) + x), in place on d_out
__global__ void k_final(float* __restrict__ z, const float* __restrict__ x,
                        const float* __restrict__ coef) {
  size_t i = (size_t)blockIdx.x * blockDim.x + threadIdx.x;   // covers NB*NO*NTV
  int o = (int)((i >> 13) & 255);
  float v = coef[o * 2] * z[i] + coef[o * 2 + 1] + x[i];
  z[i] = v > 0.f ? v : 0.f;
}

// ---------------------------------------------------------------------------
extern "C" void kernel_launch(void* const* d_in, const int* in_sizes, int n_in,
                              void* d_out, int out_size, void* d_ws, size_t ws_size,
                              hipStream_t stream) {
  const float* x       = (const float*)d_in[0];
  const float* A_param = (const float*)d_in[1];
  const float* alpha   = (const float*)d_in[2];
  const float* beta    = (const float*)d_in[3];
  const float* w1      = (const float*)d_in[4];
  const float* b1      = (const float*)d_in[5];
  const float* w2      = (const float*)d_in[6];
  const float* b2      = (const float*)d_in[7];
  const float* stem_w  = (const float*)d_in[8];
  const float* stem_b  = (const float*)d_in[9];
  const float* stem_g  = (const float*)d_in[10];
  const float* stem_be = (const float*)d_in[11];
  const float* head_w  = (const float*)d_in[12];
  const float* head_b  = (const float*)d_in[13];
  const float* head_g  = (const float*)d_in[14];
  const float* head_be = (const float*)d_in[15];
  float* out = (float*)d_out;

  // Workspace layout (float-index offsets)
  float* ws = (float*)d_ws;
  float* tp       = ws;                              // 262144
  float* x1       = ws + 262144;                     // 262144
  float* x2       = ws + 524288;                     // 262144
  float* ada      = ws + 786432;                     // 524288
  float* partial  = ws + 1310720;                    // 8192
  float* stemcoef = ws + 1318912;                    // 512
  float* headcoef = ws + 1319424;                    // 512
  unsigned short* WstemU = (unsigned short*)(ws + 1320960);  // 65536 u16
  unsigned short* WheadU = (unsigned short*)(ws + 1353728);  // 65536 u16
  unsigned short* YstU   = (unsigned short*)(ws + 2097152);  // 33554432 u16
  unsigned short* H2U    = (unsigned short*)(ws + 18874368); // 33554432 u16
  (void)ws_size; (void)n_in; (void)in_sizes; (void)out_size;

  // 0. weights -> bf16 (one-time per call; feeds async LDS copies)
  k_w2bf<<<(NO * NC + 255) / 256, 256, 0, stream>>>(stem_w, WstemU, NO * NC);
  k_w2bf<<<(NO * NO + 255) / 256, 256, 0, stream>>>(head_w, WheadU, NO * NO);
  // 1. temporal mean
  k_tp<<<NB * NC, 64, 0, stream>>>(x, tp);
  // 2. x1 / x2 projections
  k_x1x2<<<(NB * NO * NV) / 256, 256, 0, stream>>>(tp, w1, b1, w2, b2, x1, x2);
  // 3. adaptive adjacency (softmax)
  k_ada<<<NB * NK, 256, 0, stream>>>(x1, x2, ada);
  // 4. stem conv1x1 (WMMA bf16, f32 input, bf16 output)
  k_gemm256<false, true><<<dim3(NTV / 128, NO / 128, NB), 256, 0, stream>>>(
      WstemU, stem_b, x, YstU);
  // 5-6. stem BN stats (deterministic two-stage)
  k_chan_partial_bf16<<<NB * NO, 256, 0, stream>>>(YstU, partial);
  k_chan_final<<<1, NO, 0, stream>>>(partial, stem_g, stem_be, stemcoef);
  // 7. fused BN+ReLU + A-build + graph einsum (WMMA bf16)
  k_einsum<<<NB * NK * NCB, 256, 0, stream>>>(YstU, stemcoef, A_param, x1, x2,
                                              ada, alpha, beta, H2U);
  // 8. head conv1x1 (WMMA bf16, bf16 input, f32 output) -> z into d_out
  k_gemm256<true, false><<<dim3(NTV / 128, NO / 128, NB), 256, 0, stream>>>(
      WheadU, head_b, H2U, out);
  // 9-10. head BN stats
  k_chan_partial_f32<<<NB * NO, 256, 0, stream>>>(out, partial);
  k_chan_final<<<1, NO, 0, stream>>>(partial, head_g, head_be, headcoef);
  // 11. final BN + residual + ReLU (in place)
  k_final<<<(NB * NO * NTV) / 256, 256, 0, stream>>>(out, x, headcoef);
}
